// Generator_48687749267693
// MI455X (gfx1250) — matrix operand order repaired and checked by reading, not compiled, not run on hardware
//
#include <hip/hip_runtime.h>
#include <hip/hip_bf16.h>
#include <math.h>

typedef __attribute__((ext_vector_type(16))) __bf16 v16bf;
typedef __attribute__((ext_vector_type(8)))  float  v8f;

#define BATCH 8
#define CIN   512
#define COUT  512
#define SDIM  512
#define HH    32
#define YS    65     // convT output spatial size (2*32+1)
#define OS    64     // final output spatial size

// workspace layout (bytes)
#define S_OFF   0
#define WA_OFF  16384
#define WA_BYTES ((size_t)BATCH * 9 * COUT * CIN * 2)      // 37.75 MB bf16, WMMA-A swizzled
#define XT_OFF  (WA_OFF + WA_BYTES)
#define XT_BYTES ((size_t)BATCH * 34 * 34 * CIN * 2)       // 9.47 MB bf16, channel-last + halo
#define Y_OFF   (XT_OFF + XT_BYTES)

// ---------------------------------------------------------------------------
// 1) style modulation: s[b][cin] = dot(style[b], mod_w[cin]) * lin_scale + mod_b[cin]
__global__ __launch_bounds__(256)
void style_linear(const float* __restrict__ style, const float* __restrict__ mod_w,
                  const float* __restrict__ mod_b, float* __restrict__ s)
{
    int i = blockIdx.x * 256 + threadIdx.x;
    if (i >= BATCH * CIN) return;
    int b = i / CIN, cin = i % CIN;
    const float lin_scale = rsqrtf((float)SDIM);
    const float* st = style + (size_t)b * SDIM;
    const float* mw = mod_w + (size_t)cin * SDIM;
    float acc = 0.f;
    for (int k = 0; k < SDIM; ++k) acc += st[k] * mw[k];
    s[i] = acc * lin_scale + mod_b[cin];
}

// ---------------------------------------------------------------------------
// 2) modulate + demod weights; store bf16 in WMMA A-fragment swizzled layout:
//    wA[b][tap][mt(32)][kc(16)][lane(32)][e(16)]
//    lane = kh*16 + (cout&15);  e within v16bf = k32 & 15 (contiguous: k32 = 16*kh + e)
__global__ __launch_bounds__(256)
void modulate(const float* __restrict__ w, const float* __restrict__ s,
              __bf16* __restrict__ wA)
{
    int blk  = blockIdx.x;             // b*COUT + cout
    int b    = blk / COUT;
    int cout = blk % COUT;
    const float conv_scale = rsqrtf((float)(CIN * 9));

    __shared__ float red[256];
    __shared__ float sh_s[CIN];
    for (int i = threadIdx.x; i < CIN; i += 256) sh_s[i] = s[(size_t)b * CIN + i];
    __syncthreads();

    const float* wrow = w + (size_t)cout * CIN * 9;   // [cin][ky][kx]
    float sum = 0.f;
    for (int idx = threadIdx.x; idx < CIN * 9; idx += 256) {
        int cin = idx / 9;
        float v = conv_scale * wrow[idx] * sh_s[cin];
        sum += v * v;
    }
    red[threadIdx.x] = sum;
    __syncthreads();
    for (int s2 = 128; s2 > 0; s2 >>= 1) {
        if (threadIdx.x < s2) red[threadIdx.x] += red[threadIdx.x + s2];
        __syncthreads();
    }
    float demod = rsqrtf(red[0] + 1e-8f);

    int mt = cout >> 4, mrow = cout & 15;
    for (int idx = threadIdx.x; idx < CIN * 9; idx += 256) {
        int cin = idx / 9;
        int t   = idx % 9;                 // tap = ky*3+kx (unflipped; mapping handles convT)
        float v = conv_scale * wrow[cin * 9 + t] * sh_s[cin] * demod;
        int k32  = cin & 31, kc = cin >> 5;
        int kh   = k32 >> 4;
        int e    = k32 & 15;   // contiguous: half kh holds K(16*kh .. 16*kh+15)
        int lane = kh * 16 + mrow;
        size_t off = (((((size_t)b * 9 + t) * 32 + mt) * 16 + kc) * 32 + lane) * 16 + e;
        wA[off] = (__bf16)v;
    }
}

// ---------------------------------------------------------------------------
// 3) x -> channel-last bf16 with 1-pixel zero halo: xt[b][iy+1][ix+1][cin], 34x34
__global__ __launch_bounds__(256)
void x_transpose(const float* __restrict__ x, __bf16* __restrict__ xt)
{
    size_t i = (size_t)blockIdx.x * 256 + threadIdx.x;
    size_t total = (size_t)BATCH * 34 * 34 * CIN;
    if (i >= total) return;
    int cin = (int)(i & (CIN - 1));
    size_t r = i >> 9;
    int ixp = (int)(r % 34); r /= 34;
    int iyp = (int)(r % 34);
    int b   = (int)(r / 34);
    float v = 0.f;
    if (ixp >= 1 && ixp <= HH && iyp >= 1 && iyp <= HH)
        v = x[(((size_t)b * CIN + cin) * HH + (iyp - 1)) * HH + (ixp - 1)];
    xt[i] = (__bf16)v;
}

// ---------------------------------------------------------------------------
// 4) implicit-GEMM transposed conv via WMMA, parity-class decomposition with
//    N-register-blocking: one block covers a FULL class row (NS n-tiles of 16
//    pixels), 8 waves x 4 M-tiles = 512 couts. Per K-chunk: 4 A + NS B loads
//    feed 4*NS WMMAs. QE (column parity) is compile-time so NS and the kx tap
//    set are static and acc[4][NS] stays fully register-resident.
//    grid: (33 rows, 2 row-parities, BATCH); block 256 = 8 waves.
template <int QE>
__global__ __launch_bounds__(256)
void convt_gemm(const __bf16* __restrict__ wA, const __bf16* __restrict__ xt,
                float* __restrict__ y)
{
    constexpr int NCOLS = QE ? 32 : 33;
    constexpr int NS    = QE ? 2 : 3;     // n-tiles of 16 class-columns
    constexpr int NKX   = QE ? 1 : 2;

    int row = blockIdx.x;
    int pe  = blockIdx.y;                 // row parity
    int b   = blockIdx.z;
    int nrows = pe ? 32 : 33;
    if (row >= nrows) return;             // block-uniform: EXEC stays full for WMMA
    int p = pe + 2 * row;                 // y row, 0..64

    int lane = threadIdx.x & 31;
    int wv   = threadIdx.x >> 5;
    int mt0  = wv * 4;                    // this wave's first M-tile (of 32)
    int n    = lane & 15;                 // pixel column within an n-tile
    int kh   = lane >> 4;                 // K-half for A/B fragments

    v8f acc[4][NS];
    #pragma unroll
    for (int m = 0; m < 4; ++m)
        #pragma unroll
        for (int j = 0; j < NS; ++j)
            acc[m][j] = (v8f){0.f,0.f,0.f,0.f,0.f,0.f,0.f,0.f};

    // row-parity taps (runtime), column-parity taps (compile-time)
    int kys[2], nky;
    if (pe) { kys[0] = 1; nky = 1; } else { kys[0] = 0; kys[1] = 2; nky = 2; }

    for (int a = 0; a < nky; ++a) {
        int ky  = kys[a];
        int iyp = ((p - ky) >> 1) + 1;               // -1..32 -> 0..33 (zero halo)
        #pragma unroll
        for (int c = 0; c < NKX; ++c) {
            int kx = QE ? 1 : 2 * c;
            int t  = ky * 3 + kx;
            // per-n-tile x base pointers (clamped into zero halo; garbage cols masked at store)
            const __bf16* xb[NS];
            #pragma unroll
            for (int j = 0; j < NS; ++j) {
                int qq  = QE + 2 * (j * 16 + n);
                int ixp = ((qq - kx) >> 1) + 1;
                if (ixp < 0)  ixp = 0;
                if (ixp > 33) ixp = 33;
                xb[j] = xt + (((size_t)b * 34 + iyp) * 34 + ixp) * CIN + kh * 16;
            }
            const __bf16* wt = wA + (size_t)(b * 9 + t) * ((size_t)COUT * CIN);
            for (int kc = 0; kc < 16; ++kc) {
                v16bf bf[NS];
                #pragma unroll
                for (int j = 0; j < NS; ++j)
                    bf[j] = *(const v16bf*)(xb[j] + kc * 32);
                #pragma unroll
                for (int m = 0; m < 4; ++m) {
                    v16bf af = *(const v16bf*)(wt + ((size_t)(mt0 + m) * 16 + kc) * (32 * 16)
                                                  + (size_t)lane * 16);
                    #pragma unroll
                    for (int j = 0; j < NS; ++j)
                        acc[m][j] = __builtin_amdgcn_wmma_f32_16x16x32_bf16(
                            false, af, false, bf[j], (short)0, acc[m][j], false, false);
                }
            }
        }
    }

    // D layout: VGPR r -> (M = r + kh*8, N = lane&15); mask tail columns
    #pragma unroll
    for (int j = 0; j < NS; ++j) {
        int ccol = j * 16 + n;
        if (ccol < NCOLS) {
            int q = QE + 2 * ccol;
            #pragma unroll
            for (int m = 0; m < 4; ++m) {
                int cout0 = (mt0 + m) * 16;
                float* yb = y + (((size_t)b * COUT + cout0) * YS + p) * YS + q;
                #pragma unroll
                for (int r = 0; r < 8; ++r) {
                    int mloc = r + kh * 8;
                    yb[(size_t)mloc * YS * YS] = acc[m][j][r];
                }
            }
        }
    }
}

// ---------------------------------------------------------------------------
// 5) depthwise 4x4 blur (pad 1) + bias + leaky-relu * sqrt(2)
__global__ __launch_bounds__(256)
void blur_act(const float* __restrict__ y, const float* __restrict__ bias,
              float* __restrict__ out)
{
    size_t i = (size_t)blockIdx.x * 256 + threadIdx.x;
    if (i >= (size_t)BATCH * COUT * OS * OS) return;
    int ox = (int)(i & 63);
    size_t r = i >> 6;
    int oy = (int)(r & 63); r >>= 6;
    int o  = (int)(r & (COUT - 1));
    int b  = (int)(r >> 9);

    const float c[4] = {1.f, 3.f, 3.f, 1.f};
    const float* yb = y + ((size_t)b * COUT + o) * (YS * YS);
    float acc = 0.f;
    #pragma unroll
    for (int ii = 0; ii < 4; ++ii) {
        int py = oy - 1 + ii;
        if (py < 0 || py > 64) continue;
        #pragma unroll
        for (int jj = 0; jj < 4; ++jj) {
            int px = ox - 1 + jj;
            if (px < 0 || px > 64) continue;
            acc += c[ii] * c[jj] * yb[py * YS + px];
        }
    }
    acc = acc * (1.0f / 16.0f) + bias[o];
    out[i] = (acc >= 0.f ? acc : 0.2f * acc) * 1.41421356237309515f;
}

// ---------------------------------------------------------------------------
extern "C" void kernel_launch(void* const* d_in, const int* in_sizes, int n_in,
                              void* d_out, int out_size, void* d_ws, size_t ws_size,
                              hipStream_t stream)
{
    (void)in_sizes; (void)n_in; (void)out_size; (void)ws_size;
    const float* x      = (const float*)d_in[0];
    const float* style  = (const float*)d_in[1];
    const float* weight = (const float*)d_in[2];
    const float* mod_w  = (const float*)d_in[3];
    const float* mod_b  = (const float*)d_in[4];
    const float* bias   = (const float*)d_in[5];
    float* out = (float*)d_out;

    char* ws = (char*)d_ws;
    float*  s  = (float*)(ws + S_OFF);
    __bf16* wA = (__bf16*)(ws + WA_OFF);
    __bf16* xt = (__bf16*)(ws + XT_OFF);
    float*  y  = (float*)(ws + Y_OFF);

    style_linear<<<(BATCH * CIN + 255) / 256, 256, 0, stream>>>(style, mod_w, mod_b, s);

    {
        size_t total = (size_t)BATCH * 34 * 34 * CIN;
        x_transpose<<<(unsigned)((total + 255) / 256), 256, 0, stream>>>(x, xt);
    }

    modulate<<<BATCH * COUT, 256, 0, stream>>>(weight, s, wA);

    convt_gemm<0><<<dim3(33, 2, BATCH), 256, 0, stream>>>(wA, xt, y);  // even columns
    convt_gemm<1><<<dim3(33, 2, BATCH), 256, 0, stream>>>(wA, xt, y);  // odd columns

    {
        size_t total = (size_t)BATCH * COUT * OS * OS;
        blur_act<<<(unsigned)(total / 256), 256, 0, stream>>>(y, bias, out);
    }
}
